// CausalSelfAttention_26980984554238
// MI455X (gfx1250) — matrix operand order repaired and checked by reading, not compiled, not run on hardware
//
#include <hip/hip_runtime.h>
#include <hip/hip_bf16.h>
#include <math.h>
#include <stdint.h>

#define D_MODEL 1024
#define NHEAD   16
#define DK      64
#define TSEQ    2048
#define BATCH   4
#define MROWS   (BATCH * TSEQ)   // 8192

typedef __attribute__((ext_vector_type(16))) __bf16 v16bf;
typedef __attribute__((ext_vector_type(8)))  float  v8f;

union ABfrag {
    v16bf v;
    uint4 q[2];
    unsigned short h[16];
};

__device__ __forceinline__ unsigned short f2bf(float f) {
    unsigned int u = __float_as_uint(f);
    u += 0x7FFFu + ((u >> 16) & 1u);   // round-to-nearest-even
    return (unsigned short)(u >> 16);
}
__device__ __forceinline__ unsigned int pack2bf(float a, float b) {
    return (unsigned int)f2bf(a) | ((unsigned int)f2bf(b) << 16);
}

__device__ __forceinline__ v8f wmma_bf16(const ABfrag& a, const ABfrag& b, v8f c) {
    return __builtin_amdgcn_wmma_f32_16x16x32_bf16(false, a.v, false, b.v, (short)0, c, false, false);
}

// A operand: 16(M) x 32(K) bf16 from row-major [*, ld] (K contiguous).
__device__ __forceinline__ void load_frag_A(ABfrag& f, const unsigned short* base,
                                            int row, int ld, int k0, int lane) {
    int m  = row + (lane & 15);
    int k1 = k0 + ((lane < 16) ? 0 : 8);
    const unsigned short* p = base + m * ld + k1;
    f.q[0] = *reinterpret_cast<const uint4*>(p);
    f.q[1] = *reinterpret_cast<const uint4*>(p + 16);
}

// B operand: 32(K) x 16(N) bf16; source stored row-major [N, ld] (K contiguous per N row).
__device__ __forceinline__ void load_frag_B(ABfrag& f, const unsigned short* base,
                                            int col, int ld, int k0, int lane) {
    int n  = col + (lane & 15);
    int kb = k0 + ((lane < 16) ? 0 : 8);
    const unsigned short* p = base + n * ld + kb;
    f.q[0] = *reinterpret_cast<const uint4*>(p);
    f.q[1] = *reinterpret_cast<const uint4*>(p + 16);
}

// CDNA5 async global->LDS copy (ASYNCcnt-tracked). lds byte offset = low 32 bits
// of the generic pointer (flat LDS aperture keeps the offset in addr[31:0]).
__device__ __forceinline__ void async_ld16(const unsigned short* gsrc, unsigned short* ldst) {
    unsigned loff = (unsigned)(uintptr_t)ldst;
    asm volatile("global_load_async_to_lds_b128 %0, %1, off"
                 :: "v"(loff), "v"(gsrc)
                 : "memory");
}

// ---------------- fp32 -> bf16 conversion (vectorized) ----------------
__global__ void cvt_f32_bf16_v4(const float4* __restrict__ src,
                                unsigned short* __restrict__ dst, int n4) {
    int i = blockIdx.x * blockDim.x + threadIdx.x;
    int stride = gridDim.x * blockDim.x;
    uint2* d2 = reinterpret_cast<uint2*>(dst);
    for (; i < n4; i += stride) {
        float4 v = src[i];
        uint2 o;
        o.x = pack2bf(v.x, v.y);
        o.y = pack2bf(v.z, v.w);
        d2[i] = o;
    }
}

// ---------------- QKV projection GEMM ----------------
// C[8192, 3072] = xbf[8192,1024] @ Wqkv[3072,1024]^T + bqkv
// scatter: Q (scaled by 1/8) and K -> [B,H,T,dk], V -> [B,H,dk,T] (transposed), bf16
__global__ void __launch_bounds__(256) qkv_gemm(const unsigned short* __restrict__ xbf,
                                                const unsigned short* __restrict__ wbf,
                                                const float* __restrict__ bqkv,
                                                unsigned short* __restrict__ Qbf,
                                                unsigned short* __restrict__ Kbf,
                                                unsigned short* __restrict__ Vbf) {
    int lane  = threadIdx.x & 31;
    int w     = threadIdx.x >> 5;                 // 8 waves: 2 M-groups x 4 N-groups
    int mbase = blockIdx.x * 64 + (w & 1) * 32;
    int nbase = blockIdx.y * 256 + (w >> 1) * 64;

    v8f zero = {0.f,0.f,0.f,0.f,0.f,0.f,0.f,0.f};
    v8f c[2][4] = {{zero, zero, zero, zero}, {zero, zero, zero, zero}};

    for (int k0 = 0; k0 < D_MODEL; k0 += 32) {
        // preload all fragments into distinct registers -> staggered loadcnt waits
        ABfrag a0, a1, b0, b1, b2, b3;
        load_frag_A(a0, xbf, mbase,      D_MODEL, k0, lane);
        load_frag_A(a1, xbf, mbase + 16, D_MODEL, k0, lane);
        load_frag_B(b0, wbf, nbase,      D_MODEL, k0, lane);
        load_frag_B(b1, wbf, nbase + 16, D_MODEL, k0, lane);
        load_frag_B(b2, wbf, nbase + 32, D_MODEL, k0, lane);
        load_frag_B(b3, wbf, nbase + 48, D_MODEL, k0, lane);
        __builtin_prefetch(xbf + (size_t)(mbase + (lane & 15)) * D_MODEL + k0 + 128, 0, 3);
        __builtin_prefetch(wbf + (size_t)(nbase + (lane & 15)) * D_MODEL + k0 + 128, 0, 3);
        c[0][0] = wmma_bf16(a0, b0, c[0][0]);
        c[1][0] = wmma_bf16(a1, b0, c[1][0]);
        c[0][1] = wmma_bf16(a0, b1, c[0][1]);
        c[1][1] = wmma_bf16(a1, b1, c[1][1]);
        c[0][2] = wmma_bf16(a0, b2, c[0][2]);
        c[1][2] = wmma_bf16(a1, b2, c[1][2]);
        c[0][3] = wmma_bf16(a0, b3, c[0][3]);
        c[1][3] = wmma_bf16(a1, b3, c[1][3]);
    }

    int colLane = lane & 15;
    int rowOff  = (lane < 16) ? 0 : 8;
#pragma unroll
    for (int g = 0; g < 2; g++) {
#pragma unroll
        for (int f = 0; f < 4; f++) {
            int e     = nbase + f * 16 + colLane;     // 0..3071
            float bias = bqkv[e];
            int which = e >> 10;                      // 0=q 1=k 2=v
            int fe    = e & 1023;
            int hh    = fe >> 6;
            int dd    = fe & 63;
#pragma unroll
            for (int r = 0; r < 8; r++) {
                int mg = mbase + g * 16 + rowOff + r;
                int bb = mg >> 11;
                int tt = mg & 2047;
                float val = c[g][f][r] + bias;
                int bh = bb * NHEAD + hh;
                if (which == 0)
                    Qbf[(bh * TSEQ + tt) * DK + dd] = f2bf(val * 0.125f); // 1/sqrt(64)
                else if (which == 1)
                    Kbf[(bh * TSEQ + tt) * DK + dd] = f2bf(val);
                else
                    Vbf[(bh * DK + dd) * TSEQ + tt] = f2bf(val);
            }
        }
    }
}

// ---------------- flash attention (causal, online softmax) ----------------
// grid: B*H*(T/64) blocks, 128 threads = 4 waves, each wave owns a 16-row q tile.
// K/V staged per-block in LDS via async global->LDS copies, double-buffered.
__global__ void __launch_bounds__(128) flash_attn(const unsigned short* __restrict__ Qbf,
                                                  const unsigned short* __restrict__ Kbf,
                                                  const unsigned short* __restrict__ Vbf,
                                                  unsigned short* __restrict__ attn) {
    __shared__ __align__(16) unsigned short kls[2][32 * DK];   // [key][d]
    __shared__ __align__(16) unsigned short vls[2][DK * 32];   // [d][key]
    __shared__ __align__(16) unsigned short pbuf[4][16 * 32];

    int lane = threadIdx.x & 31;
    int w    = threadIdx.x >> 5;
    int bid  = blockIdx.x;
    int qc   = bid & 31;          // T/64 = 32 chunks
    int bh   = bid >> 5;          // b*16 + h
    int q0   = qc * 64 + w * 16;

    const unsigned short* Qb = Qbf + (size_t)bh * TSEQ * DK;
    const unsigned short* Kb = Kbf + (size_t)bh * TSEQ * DK;
    const unsigned short* Vb = Vbf + (size_t)bh * DK * TSEQ;

    ABfrag qf0, qf1;
    load_frag_A(qf0, Qb, q0, DK, 0, lane);
    load_frag_A(qf1, Qb, q0, DK, 32, lane);

    v8f zero = {0.f,0.f,0.f,0.f,0.f,0.f,0.f,0.f};
    v8f o[4] = {zero, zero, zero, zero};
    float m_s[8], l_s[8];
#pragma unroll
    for (int r = 0; r < 8; r++) { m_s[r] = -3.0e38f; l_s[r] = 0.f; }

    int colLane = lane & 15;
    int rowOff  = (lane < 16) ? 0 : 8;
    unsigned short* pl = pbuf[w];
    int nkb = 2 * qc + 2;   // key blocks of 32 covering the whole 64-row chunk

    // cooperative async stage of one 32-key K/V block (4 b128 per wave = 8KB/block)
    auto issue_kv = [&](int bsel, int kbase) {
#pragma unroll
        for (int i = 0; i < 2; i++) {
            int cch = w * 64 + i * 32 + lane;     // 256 chunks of 16B, K tile
            int row = cch >> 3, koff = (cch & 7) * 8;
            async_ld16(Kb + (size_t)(kbase + row) * DK + koff, &kls[bsel][row * DK + koff]);
        }
#pragma unroll
        for (int i = 0; i < 2; i++) {
            int cch = w * 64 + i * 32 + lane;     // 256 chunks of 16B, V tile
            int dd = cch >> 2, koff = (cch & 3) * 8;
            async_ld16(Vb + (size_t)dd * TSEQ + kbase + koff, &vls[bsel][dd * 32 + koff]);
        }
    };

    issue_kv(0, 0);
    for (int kb = 0; kb < nkb; kb++) {
        int kbase = kb * 32;
        if (kb + 1 < nkb) {
            issue_kv((kb + 1) & 1, (kb + 1) * 32);
            asm volatile("s_wait_asynccnt 4" ::: "memory");   // current block landed
        } else {
            asm volatile("s_wait_asynccnt 0" ::: "memory");
        }
        __syncthreads();
        const unsigned short* kc = kls[kb & 1];
        const unsigned short* vc = vls[kb & 1];

        if (kbase <= q0 + 15) {   // wave-uniform: this wave still has unmasked keys
            // ---- S = Q @ K^T (two 16x16 tiles, K-dim = dk = 64), K from LDS ----
            v8f s0 = zero, s1 = zero;
            {
                ABfrag bk0, bk1, bk2, bk3;
                load_frag_B(bk0, kc, 0,  DK, 0,  lane);
                load_frag_B(bk1, kc, 0,  DK, 32, lane);
                load_frag_B(bk2, kc, 16, DK, 0,  lane);
                load_frag_B(bk3, kc, 16, DK, 32, lane);
                s0 = wmma_bf16(qf0, bk0, s0);
                s0 = wmma_bf16(qf1, bk1, s0);
                s1 = wmma_bf16(qf0, bk2, s1);
                s1 = wmma_bf16(qf1, bk3, s1);
            }
            // ---- online softmax ----
            float p0a[8], p1a[8], alpha[8];
#pragma unroll
            for (int r = 0; r < 8; r++) {
                int rowg = q0 + rowOff + r;
                int c0 = kbase + colLane;
                int c1 = c0 + 16;
                float s0v = (c0 <= rowg) ? s0[r] : -3.0e38f;
                float s1v = (c1 <= rowg) ? s1[r] : -3.0e38f;
                float mx = fmaxf(s0v, s1v);
                mx = fmaxf(mx, __shfl_xor(mx, 1, 32));
                mx = fmaxf(mx, __shfl_xor(mx, 2, 32));
                mx = fmaxf(mx, __shfl_xor(mx, 4, 32));
                mx = fmaxf(mx, __shfl_xor(mx, 8, 32));
                float mn = fmaxf(m_s[r], mx);
                float e0 = __expf(s0v - mn);
                float e1 = __expf(s1v - mn);
                float rs = e0 + e1;
                rs += __shfl_xor(rs, 1, 32);
                rs += __shfl_xor(rs, 2, 32);
                rs += __shfl_xor(rs, 4, 32);
                rs += __shfl_xor(rs, 8, 32);
                float al = __expf(m_s[r] - mn);
                l_s[r] = l_s[r] * al + rs;
                m_s[r] = mn;
                alpha[r] = al; p0a[r] = e0; p1a[r] = e1;
            }
#pragma unroll
            for (int f = 0; f < 4; f++)
#pragma unroll
                for (int r = 0; r < 8; r++) o[f][r] *= alpha[r];

            // ---- P (C layout, f32) -> LDS col-major bf16 [col*16 + row] ----
            {
                uint4 v0, v1;
                v0.x = pack2bf(p0a[0], p0a[1]); v0.y = pack2bf(p0a[2], p0a[3]);
                v0.z = pack2bf(p0a[4], p0a[5]); v0.w = pack2bf(p0a[6], p0a[7]);
                v1.x = pack2bf(p1a[0], p1a[1]); v1.y = pack2bf(p1a[2], p1a[3]);
                v1.z = pack2bf(p1a[4], p1a[5]); v1.w = pack2bf(p1a[6], p1a[7]);
                *reinterpret_cast<uint4*>(pl + colLane * 16 + rowOff)        = v0;
                *reinterpret_cast<uint4*>(pl + (colLane + 16) * 16 + rowOff) = v1;
            }
            // ---- reload P as WMMA A fragment (16x32) ----
            ABfrag pf;
            {
                int prow = colLane;
                int pk   = (lane < 16) ? 0 : 8;
#pragma unroll
                for (int e = 0; e < 8; e++) pf.h[e]     = pl[(pk + e) * 16 + prow];
#pragma unroll
                for (int e = 0; e < 8; e++) pf.h[8 + e] = pl[(pk + 16 + e) * 16 + prow];
            }
            // ---- O += P @ V (V tile [d][key] in LDS -> direct B fragments) ----
            {
                ABfrag vf0, vf1, vf2, vf3;
                load_frag_B(vf0, vc, 0,  32, 0, lane);
                load_frag_B(vf1, vc, 16, 32, 0, lane);
                load_frag_B(vf2, vc, 32, 32, 0, lane);
                load_frag_B(vf3, vc, 48, 32, 0, lane);
                o[0] = wmma_bf16(pf, vf0, o[0]);
                o[1] = wmma_bf16(pf, vf1, o[1]);
                o[2] = wmma_bf16(pf, vf2, o[2]);
                o[3] = wmma_bf16(pf, vf3, o[3]);
            }
        }
        __syncthreads();
    }

    // ---- normalize and store attn_out bf16 [B, T, D] ----
    float inv[8];
#pragma unroll
    for (int r = 0; r < 8; r++) inv[r] = 1.0f / l_s[r];
    int b_idx = bh >> 4;
    int h_idx = bh & 15;
#pragma unroll
    for (int f = 0; f < 4; f++) {
        int dcol = h_idx * 64 + f * 16 + colLane;
#pragma unroll
        for (int r = 0; r < 8; r++) {
            int t = q0 + rowOff + r;
            attn[((size_t)b_idx * TSEQ + t) * D_MODEL + dcol] = f2bf(o[f][r] * inv[r]);
        }
    }
}

// ---------------- output projection GEMM ----------------
// out[8192,1024] = attn[8192,1024] @ Wo[1024,1024]^T + bo   (fp32 out)
__global__ void __launch_bounds__(256) out_gemm(const unsigned short* __restrict__ abf,
                                                const unsigned short* __restrict__ wobf,
                                                const float* __restrict__ bo,
                                                float* __restrict__ out) {
    int lane  = threadIdx.x & 31;
    int w     = threadIdx.x >> 5;
    int mbase = blockIdx.x * 64 + (w & 1) * 32;
    int nbase = blockIdx.y * 256 + (w >> 1) * 64;

    v8f zero = {0.f,0.f,0.f,0.f,0.f,0.f,0.f,0.f};
    v8f c[2][4] = {{zero, zero, zero, zero}, {zero, zero, zero, zero}};

    for (int k0 = 0; k0 < D_MODEL; k0 += 32) {
        ABfrag a0, a1, b0, b1, b2, b3;
        load_frag_A(a0, abf, mbase,      D_MODEL, k0, lane);
        load_frag_A(a1, abf, mbase + 16, D_MODEL, k0, lane);
        load_frag_B(b0, wobf, nbase,      D_MODEL, k0, lane);
        load_frag_B(b1, wobf, nbase + 16, D_MODEL, k0, lane);
        load_frag_B(b2, wobf, nbase + 32, D_MODEL, k0, lane);
        load_frag_B(b3, wobf, nbase + 48, D_MODEL, k0, lane);
        __builtin_prefetch(abf + (size_t)(mbase + (lane & 15)) * D_MODEL + k0 + 128, 0, 3);
        __builtin_prefetch(wobf + (size_t)(nbase + (lane & 15)) * D_MODEL + k0 + 128, 0, 3);
        c[0][0] = wmma_bf16(a0, b0, c[0][0]);
        c[1][0] = wmma_bf16(a1, b0, c[1][0]);
        c[0][1] = wmma_bf16(a0, b1, c[0][1]);
        c[1][1] = wmma_bf16(a1, b1, c[1][1]);
        c[0][2] = wmma_bf16(a0, b2, c[0][2]);
        c[1][2] = wmma_bf16(a1, b2, c[1][2]);
        c[0][3] = wmma_bf16(a0, b3, c[0][3]);
        c[1][3] = wmma_bf16(a1, b3, c[1][3]);
    }

    int colLane = lane & 15;
    int rowOff  = (lane < 16) ? 0 : 8;
#pragma unroll
    for (int g = 0; g < 2; g++) {
#pragma unroll
        for (int f = 0; f < 4; f++) {
            int e = nbase + f * 16 + colLane;
            float bias = bo[e];
#pragma unroll
            for (int r = 0; r < 8; r++) {
                int mg = mbase + g * 16 + rowOff + r;
                out[(size_t)mg * D_MODEL + e] = c[g][f][r] + bias;
            }
        }
    }
}

extern "C" void kernel_launch(void* const* d_in, const int* in_sizes, int n_in,
                              void* d_out, int out_size, void* d_ws, size_t ws_size,
                              hipStream_t stream) {
    const float* x    = (const float*)d_in[0];
    const float* Wqkv = (const float*)d_in[1];
    const float* bqkv = (const float*)d_in[2];
    const float* Wo   = (const float*)d_in[3];
    const float* bo   = (const float*)d_in[4];
    float* out = (float*)d_out;

    char* ws = (char*)d_ws;
    size_t off = 0;
    auto alloc = [&](size_t bytes) -> void* {
        void* p = ws + off;
        off += (bytes + 255) & ~(size_t)255;
        return p;
    };
    unsigned short* xbf  = (unsigned short*)alloc((size_t)MROWS * D_MODEL * 2);
    unsigned short* wqbf = (unsigned short*)alloc((size_t)3 * D_MODEL * D_MODEL * 2);
    unsigned short* wobf = (unsigned short*)alloc((size_t)D_MODEL * D_MODEL * 2);
    unsigned short* qbf  = (unsigned short*)alloc((size_t)BATCH * NHEAD * TSEQ * DK * 2);
    unsigned short* kbf  = (unsigned short*)alloc((size_t)BATCH * NHEAD * TSEQ * DK * 2);
    unsigned short* vbf  = (unsigned short*)alloc((size_t)BATCH * NHEAD * TSEQ * DK * 2);
    unsigned short* abf  = (unsigned short*)alloc((size_t)MROWS * D_MODEL * 2);
    (void)alloc(4096);   // pad so tail prefetches stay inside the workspace

    // 1) fp32 -> bf16 conversions
    {
        int n4x = MROWS * D_MODEL / 4;
        int n4q = 3 * D_MODEL * D_MODEL / 4;
        int n4o = D_MODEL * D_MODEL / 4;
        cvt_f32_bf16_v4<<<1024, 256, 0, stream>>>((const float4*)x,    xbf,  n4x);
        cvt_f32_bf16_v4<<<1024, 256, 0, stream>>>((const float4*)Wqkv, wqbf, n4q);
        cvt_f32_bf16_v4<<<1024, 256, 0, stream>>>((const float4*)Wo,   wobf, n4o);
    }
    // 2) fused QKV projection (+ bias, Q scaling, V transpose)
    qkv_gemm<<<dim3(MROWS / 64, 3 * D_MODEL / 256), 256, 0, stream>>>(
        xbf, wqbf, bqkv, qbf, kbf, vbf);
    // 3) causal flash attention (async K/V staging, double-buffered)
    flash_attn<<<BATCH * NHEAD * (TSEQ / 64), 128, 0, stream>>>(qbf, kbf, vbf, abf);
    // 4) output projection
    out_gemm<<<dim3(MROWS / 64, D_MODEL / 256), 256, 0, stream>>>(abf, wobf, bo, out);
}
